// SelfAttention_30013231464749
// MI455X (gfx1250) — matrix operand order repaired and checked by reading, not compiled, not run on hardware
//
#include <hip/hip_runtime.h>
#include <hip/hip_bf16.h>
#include <math.h>

// Problem constants (from reference)
constexpr int BATCH = 4;
constexpr int SEQ   = 4096;
constexpr int EMB   = 1024;
constexpr int DK    = 64;
constexpr float LOG2E = 1.4426950408889634f;

typedef __attribute__((ext_vector_type(16))) __bf16 v16bf;
typedef __attribute__((ext_vector_type(8)))  float  v8f;

union Frag16 { v16bf v; unsigned int u[8]; };

__device__ __forceinline__ v8f wmma_bf16(v16bf a, v16bf b, v8f c) {
  // D = A(16x32 bf16) * B(32x16 bf16) + C(16x16 f32)
  return __builtin_amdgcn_wmma_f32_16x16x32_bf16(false, a, false, b, (short)0, c,
                                                 false, false);
}

// A-matrix frag (16x32 bf16): lane holds row M = lane%16; dword j of the frag
// holds K-pair {2j,2j+1} (+8 for upper lane half, +16 for j>=4).
__device__ __forceinline__ v16bf load_a_frag(const unsigned int* row, int h, int ko) {
  Frag16 f;
#pragma unroll
  for (int j = 0; j < 8; ++j)
    f.u[j] = row[ko + (j < 4 ? j : j + 4) + 4 * h];
  return f.v;
}

// B-matrix frag (32x16 bf16): lane holds column N = lane%16; dword j holds
// K-pair {2j,2j+1} (+8 for upper lane half, +16 for j>=4); K stored contiguously per column.
__device__ __forceinline__ v16bf load_b_frag(const unsigned int* row, int h, int ko) {
  Frag16 f;
#pragma unroll
  for (int j = 0; j < 8; ++j)
    f.u[j] = row[ko + (j < 4 ? j : j + 4) + 4 * h];
  return f.v;
}

// Async HBM -> LDS copy of 16 bytes per lane (GLOBAL_LOAD_ASYNC_TO_LDS_B128,
// GVS addressing: 64-bit SGPR base + 32-bit per-lane offset).  Tracked by
// ASYNCcnt; completion enforced with s_wait_asynccnt.
__device__ __forceinline__ void async_cp16(unsigned lds_addr,
                                           unsigned long long gbase,
                                           int goff) {
  asm volatile("global_load_async_to_lds_b128 %0, %1, %2"
               :: "v"(lds_addr), "v"(goff), "s"(gbase)
               : "memory");
}

// ---------------------------------------------------------------------------
// Kernel 1: Q/K/V projection.  Per block: 128 rows of X, 8 waves, one 16-row
// M-tile per wave, 12 N-tiles (Q:4, K:4, V:4) accumulated in f32 via WMMA.
// Q is pre-scaled by 1/sqrt(d_k).  V is written TRANSPOSED per batch
// (Vt[b][dk][seq]) so the attention kernel can stage V^T with contiguous
// async B128 copies.
// ---------------------------------------------------------------------------
__global__ __launch_bounds__(256) void qkv_kernel(
    const float* __restrict__ X, const float* __restrict__ Wq,
    const float* __restrict__ Wk, const float* __restrict__ Wv,
    __bf16* __restrict__ Qb, __bf16* __restrict__ Kb, __bf16* __restrict__ Vt) {
  __shared__ alignas(16) __bf16 sX[128][32];   // X tile, row-major (8 KB)
  __shared__ alignas(16) __bf16 sW[192][32];   // W tile, K-contiguous per col (12 KB)

  const int tid  = threadIdx.x;
  const int lane = tid & 31;
  const int wave = tid >> 5;
  const int half = lane >> 4;
  const int lm   = lane & 15;
  const int mbase = blockIdx.x * 128;

  v8f acc[12] = {};

  for (int k0 = 0; k0 < EMB; k0 += 32) {
    // Prefetch next X K-slab into cache (global_prefetch_b8).
    if (k0 + 32 < EMB)
      __builtin_prefetch(&X[(size_t)(mbase + (tid >> 1)) * EMB + k0 + 32 +
                            (tid & 1) * 16], 0, 0);
    // Stage X block: 128*32 = 4096 f32 -> bf16; 16 contiguous elems per thread.
#pragma unroll
    for (int i = 0; i < 16; ++i) {
      int idx = tid * 16 + i;
      int r = idx >> 5, c = idx & 31;
      sX[r][c] = (__bf16)X[(size_t)(mbase + r) * EMB + k0 + c];
    }
    // Stage W block transposed: 192 cols x 32 k = 6144 elems; 24 per thread.
#pragma unroll
    for (int i = 0; i < 24; ++i) {
      int idx = tid * 24 + i;
      int col = idx / 32, kk = idx % 32;
      const float* w = (col < 64) ? Wq : (col < 128 ? Wk : Wv);
      sW[col][kk] = (__bf16)w[(size_t)(k0 + kk) * DK + (col & 63)];
    }
    __syncthreads();

    const unsigned int* arow = (const unsigned int*)&sX[wave * 16 + lm][0];
    v16bf af = load_a_frag(arow, half, 0);
#pragma unroll
    for (int t = 0; t < 12; ++t) {
      const unsigned int* wrow = (const unsigned int*)&sW[t * 16 + lm][0];
      acc[t] = wmma_bf16(af, load_b_frag(wrow, half, 0), acc[t]);
    }
    __syncthreads();
  }

  // Store: C layout puts row (r + 8*half) in dword r, col lm.
#pragma unroll
  for (int t = 0; t < 12; ++t) {
    const float scale = (t < 4) ? 0.125f : 1.0f;  // 1/sqrt(64) folded into Q
    int col = (t & 3) * 16 + lm;
#pragma unroll
    for (int r = 0; r < 8; ++r) {
      int row = mbase + wave * 16 + r + 8 * half;
      __bf16 v = (__bf16)(acc[t][r] * scale);
      if (t < 4) {
        Qb[(size_t)row * DK + col] = v;
      } else if (t < 8) {
        Kb[(size_t)row * DK + col] = v;
      } else {
        int b = row >> 12;            // SEQ = 4096
        int rloc = row & (SEQ - 1);
        Vt[((size_t)b * DK + col) * SEQ + rloc] = v;  // transposed
      }
    }
  }
}

// ---------------------------------------------------------------------------
// Kernel 2: flash attention.  128 query rows per block (8 waves x 16); 64-key
// blocks streamed through double-buffered LDS via async B128 copies, online
// softmax, O accumulated in f32 registers.
// ---------------------------------------------------------------------------
__global__ __launch_bounds__(256) void attn_kernel(
    const __bf16* __restrict__ Qb, const __bf16* __restrict__ Kb,
    const __bf16* __restrict__ Vt, float* __restrict__ out) {
  __shared__ alignas(16) __bf16 sK[2][64][64];    // [buf][key][dk]  (16 KB)
  __shared__ alignas(16) __bf16 sVt[2][64][64];   // [buf][dk][key]  (16 KB)
  __shared__ alignas(16) __bf16 sP[8][16][64];    // per-wave P      (16 KB)

  const int tid  = threadIdx.x;
  const int lane = tid & 31;
  const int wave = tid >> 5;
  const int half = lane >> 4;
  const int lm   = lane & 15;
  const int b = blockIdx.y;
  const size_t bbase = (size_t)b * SEQ;
  const int qbase = blockIdx.x * 128 + wave * 16;

  // Uniform per-batch base addresses for the async copies.
  const unsigned long long kBase =
      (unsigned long long)(uintptr_t)(Kb + bbase * DK);
  const unsigned long long vBase =
      (unsigned long long)(uintptr_t)(Vt + (size_t)b * DK * SEQ);

  // Issue one 64-key tile (K: 8 KB contiguous; V^T: 64 rows x 128 B).
  // 4 async B128 ops per lane -> ASYNCcnt == 4 per tile per wave.
  auto issue_tile = [&](int buf, int kb) {
    const int key0 = kb * 64;
    unsigned ldsK = (unsigned)(uintptr_t)&sK[buf][0][0];
    unsigned ldsV = (unsigned)(uintptr_t)&sVt[buf][0][0];
#pragma unroll
    for (int i = 0; i < 2; ++i) {
      int c = tid + 256 * i;                       // chunk id, 512 x 16 B
      async_cp16(ldsK + c * 16, kBase, key0 * (DK * 2) + c * 16);
      int row = c >> 3;                            // dk row of V^T
      async_cp16(ldsV + c * 16, vBase,
                 row * (SEQ * 2) + key0 * 2 + (c & 7) * 16);
    }
  };

  // Q A-frags (K = d_k = 64 -> two 32-K steps), resident for whole kernel.
  v16bf qf[2];
  {
    const unsigned int* qrow =
        (const unsigned int*)(Qb + (bbase + qbase + lm) * DK);
    qf[0] = load_a_frag(qrow, half, 0);
    qf[1] = load_a_frag(qrow, half, 16);
  }

  v8f oacc[4] = {};
  float mrow[8], lrow[8];
#pragma unroll
  for (int r = 0; r < 8; ++r) { mrow[r] = -INFINITY; lrow[r] = 0.0f; }

  constexpr int NB = SEQ / 64;
  issue_tile(0, 0);

  for (int kb = 0; kb < NB; ++kb) {
    const int cur = kb & 1;
    if (kb + 1 < NB) {
      issue_tile(cur ^ 1, kb + 1);
      // Wait for the current tile (leave the 4 just-issued ops in flight).
      asm volatile("s_wait_asynccnt %0" :: "i"(4));
    } else {
      asm volatile("s_wait_asynccnt %0" :: "i"(0));
    }
    __syncthreads();  // current tile visible to all waves

    // GEMM1: S = Q * K^T  (gemm-K = d_k = 64)
    v8f sacc[4] = {};
#pragma unroll
    for (int t = 0; t < 4; ++t) {
      const unsigned int* krow = (const unsigned int*)&sK[cur][t * 16 + lm][0];
#pragma unroll
      for (int s2 = 0; s2 < 2; ++s2)
        sacc[t] = wmma_bf16(qf[s2], load_b_frag(krow, half, 16 * s2), sacc[t]);
    }

    // Online softmax: row r+8*half lives in dword r across a 16-lane group.
    float alpha[8];
#pragma unroll
    for (int r = 0; r < 8; ++r) {
      float v0 = fmaxf(fmaxf(sacc[0][r], sacc[1][r]),
                       fmaxf(sacc[2][r], sacc[3][r]));
      v0 = fmaxf(v0, __shfl_xor(v0, 1, 32));
      v0 = fmaxf(v0, __shfl_xor(v0, 2, 32));
      v0 = fmaxf(v0, __shfl_xor(v0, 4, 32));
      v0 = fmaxf(v0, __shfl_xor(v0, 8, 32));
      float mnew = fmaxf(mrow[r], v0);
      alpha[r] = exp2f((mrow[r] - mnew) * LOG2E);
      mrow[r] = mnew;
      float psum = 0.0f;
#pragma unroll
      for (int t = 0; t < 4; ++t) {
        float p = exp2f((sacc[t][r] - mnew) * LOG2E);
        sacc[t][r] = p;
        psum += p;
      }
      psum += __shfl_xor(psum, 1, 32);
      psum += __shfl_xor(psum, 2, 32);
      psum += __shfl_xor(psum, 4, 32);
      psum += __shfl_xor(psum, 8, 32);
      lrow[r] = lrow[r] * alpha[r] + psum;
    }

    // Rescale O, spill P through per-wave LDS (C layout -> A layout transpose).
#pragma unroll
    for (int t = 0; t < 4; ++t) {
#pragma unroll
      for (int r = 0; r < 8; ++r) {
        oacc[t][r] *= alpha[r];
        sP[wave][r + 8 * half][t * 16 + lm] = (__bf16)sacc[t][r];
      }
    }
    // Same-wave LDS store->load dependency: drain DScnt explicitly.
    asm volatile("s_wait_dscnt 0" ::: "memory");

    // GEMM2: O += P * V  (gemm-K = 64 keys)
    const unsigned int* prow = (const unsigned int*)&sP[wave][lm][0];
    v16bf pa0 = load_a_frag(prow, half, 0);
    v16bf pa1 = load_a_frag(prow, half, 16);
#pragma unroll
    for (int t = 0; t < 4; ++t) {
      const unsigned int* vrow = (const unsigned int*)&sVt[cur][t * 16 + lm][0];
      oacc[t] = wmma_bf16(pa0, load_b_frag(vrow, half, 0), oacc[t]);
      oacc[t] = wmma_bf16(pa1, load_b_frag(vrow, half, 16), oacc[t]);
    }
    __syncthreads();  // everyone done reading buf[cur] before it is refilled
  }

  // Epilogue: O / l, f32 out [B, S, 64].
#pragma unroll
  for (int r = 0; r < 8; ++r) {
    float inv = 1.0f / lrow[r];
    int row = qbase + r + 8 * half;
#pragma unroll
    for (int t = 0; t < 4; ++t)
      out[(bbase + row) * DK + t * 16 + lm] = oacc[t][r] * inv;
  }
}

extern "C" void kernel_launch(void* const* d_in, const int* in_sizes, int n_in,
                              void* d_out, int out_size, void* d_ws, size_t ws_size,
                              hipStream_t stream) {
  const float* X  = (const float*)d_in[0];
  const float* Wq = (const float*)d_in[1];
  const float* Wk = (const float*)d_in[2];
  const float* Wv = (const float*)d_in[3];

  // Workspace: bf16 Q, K (row-major) and V (transposed per batch), 2 MB each.
  __bf16* Qb = (__bf16*)d_ws;
  __bf16* Kb = Qb + (size_t)BATCH * SEQ * DK;
  __bf16* Vt = Kb + (size_t)BATCH * SEQ * DK;

  qkv_kernel<<<dim3((BATCH * SEQ) / 128), 256, 0, stream>>>(X, Wq, Wk, Wv, Qb, Kb, Vt);
  attn_kernel<<<dim3(SEQ / 128, BATCH), 256, 0, stream>>>(Qb, Kb, Vt, (float*)d_out);
}
